// CAM_Module_my_58059367908034
// MI455X (gfx1250) — matrix operand order repaired and checked by reading, not compiled, not run on hardware
//
#include <hip/hip_runtime.h>
#include <hip/hip_bf16.h>

typedef __attribute__((ext_vector_type(16))) _Float16 v16h;
typedef __attribute__((ext_vector_type(8)))  _Float16 h8;
typedef __attribute__((ext_vector_type(8)))  float    v8f;

static constexpr int BATCH  = 8;
static constexpr int C      = 512;
static constexpr int NPIX   = 4096;   // 64*64
static constexpr int TILE   = 128;
static constexpr int KSTEP  = 32;
static constexpr int LDSW   = 40;     // halves per LDS row (80 B, 16 B multiple)
static constexpr int KSPLIT = 4;
static constexpr int KCHUNK = NPIX / KSPLIT;   // 1024

union FragU { v16h v; h8 h[2]; };

// CDNA5 async global->LDS copy (ASYNCcnt). Low 32 bits of a generic LDS
// pointer are the wave-relative LDS offset (ISA 10.2 aperture mapping).
static __device__ __forceinline__ void cp_async_b128(void* lds, const void* g) {
  unsigned l = (unsigned)(uintptr_t)lds;
  asm volatile("global_load_async_to_lds_b128 %0, %1, off" :: "v"(l), "v"(g) : "memory");
}
static __device__ __forceinline__ void wait_async0() {
  asm volatile("s_wait_asynccnt 0x0" ::: "memory");
}

// ---------------------------------------------------------------------------
// Kernel 1: x (B,C,N) f32 -> qh (B,C,N) f16  AND  qhT (B,N,C) f16
// ---------------------------------------------------------------------------
__global__ __launch_bounds__(256) void cam_convert(
    const float* __restrict__ x, _Float16* __restrict__ qh,
    _Float16* __restrict__ qhT) {
  __shared__ float tile[32][33];
  const int b = blockIdx.z;
  const int nBase = blockIdx.x * 32;
  const int cBase = blockIdx.y * 32;
  const int tx = threadIdx.x;          // 0..31
  const int ty = threadIdx.y;          // 0..7
  const float* xb = x + (size_t)b * C * NPIX;
  _Float16* qb = qh + (size_t)b * C * NPIX;
#pragma unroll
  for (int p = 0; p < 4; ++p) {
    int r = ty + p * 8;
    float v = xb[(size_t)(cBase + r) * NPIX + nBase + tx];
    tile[r][tx] = v;
    qb[(size_t)(cBase + r) * NPIX + nBase + tx] = (_Float16)v;
  }
  __syncthreads();
  _Float16* qt = qhT + (size_t)b * NPIX * C;
#pragma unroll
  for (int p = 0; p < 4; ++p) {
    int r = ty + p * 8;
    qt[(size_t)(nBase + r) * C + cBase + tx] = (_Float16)tile[tx][r];
  }
}

// ---------------------------------------------------------------------------
// Kernel 2: split-K Gram:  part[kc,b] += qh qh^T over K-chunk kc
// grid (C/128, C/128, B*KSPLIT), block 256 (8 waves: 2 M x 4 N)
// Async double-buffered f16 tiles, one barrier per K-step.
// ---------------------------------------------------------------------------
__global__ __launch_bounds__(256) void cam_gram_wmma(
    const _Float16* __restrict__ qh, float* __restrict__ part) {
  __shared__ alignas(16) _Float16 As[2][TILE][LDSW];
  __shared__ alignas(16) _Float16 Bs[2][TILE][LDSW];

  const int b     = blockIdx.z / KSPLIT;
  const int kc    = blockIdx.z % KSPLIT;
  const int mBase = blockIdx.y * TILE;
  const int nBase = blockIdx.x * TILE;
  const int kBase = kc * KCHUNK;
  const _Float16* qb = qh + (size_t)b * C * NPIX;

  const int tid   = threadIdx.x;
  const int lane  = tid & 31;
  const int wave  = tid >> 5;
  const int waveM = wave & 1;
  const int waveN = wave >> 1;
  const int half  = lane >> 4;
  const int r15   = lane & 15;

  const int ldRow = tid >> 2;        // 64 rows/pass, 2 passes
  const int ldCol = (tid & 3) * 8;   // 8 halves = 16 B

  v8f acc[4][2] = {};

  auto issue = [&](int buf, int k0) {
#pragma unroll
    for (int p = 0; p < 2; ++p) {
      int row = ldRow + p * 64;
      cp_async_b128(&As[buf][row][ldCol], qb + (size_t)(mBase + row) * NPIX + k0 + ldCol);
      cp_async_b128(&Bs[buf][row][ldCol], qb + (size_t)(nBase + row) * NPIX + k0 + ldCol);
    }
  };

  issue(0, kBase);
  wait_async0();
  __syncthreads();

  int cur = 0;
  for (int ks = 0; ks < KCHUNK; ks += KSTEP) {
    if (ks + KSTEP < KCHUNK) issue(cur ^ 1, kBase + ks + KSTEP);

    FragU aF[4], bF[2];
#pragma unroll
    for (int i = 0; i < 4; ++i) {
      int row = waveM * 64 + i * 16 + r15;
      aF[i].h[0] = *(const h8*)(&As[cur][row][half * 8]);
      aF[i].h[1] = *(const h8*)(&As[cur][row][16 + half * 8]);
    }
#pragma unroll
    for (int j = 0; j < 2; ++j) {
      int row = waveN * 32 + j * 16 + r15;
      bF[j].h[0] = *(const h8*)(&Bs[cur][row][half * 8]);
      bF[j].h[1] = *(const h8*)(&Bs[cur][row][16 + half * 8]);
    }
#pragma unroll
    for (int i = 0; i < 4; ++i)
#pragma unroll
      for (int j = 0; j < 2; ++j)
        acc[i][j] = __builtin_amdgcn_wmma_f32_16x16x32_f16(
            false, aF[i].v, false, bF[j].v, (short)0, acc[i][j], false, false);

    wait_async0();
    __syncthreads();
    cur ^= 1;
  }

#pragma unroll
  for (int i = 0; i < 4; ++i)
#pragma unroll
    for (int j = 0; j < 2; ++j) {
      int rowG = mBase + waveM * 64 + i * 16 + half * 8;
      int colG = nBase + waveN * 32 + j * 16 + r15;
      float* dst = part + (((size_t)kc * BATCH + b) * C + rowG) * C + colG;
#pragma unroll
      for (int v = 0; v < 8; ++v) dst[(size_t)v * C] = acc[i][j][v];
    }
}

// ---------------------------------------------------------------------------
// Kernel 3: sum split-K partials + softmax(rowmax - e) = normalize(exp(min-e))
// one wave per row (row = b*C + m); output f16
// ---------------------------------------------------------------------------
__global__ __launch_bounds__(256) void cam_softmax(
    const float* __restrict__ part, _Float16* __restrict__ attn) {
  const int row  = blockIdx.x * 8 + (threadIdx.x >> 5);  // 0 .. B*C-1
  const int lane = threadIdx.x & 31;
  const size_t stride = (size_t)BATCH * C * C;           // per k-chunk
  const float* e = part + (size_t)row * C;

  float vals[16];
  float mn = 3.4e38f;
#pragma unroll
  for (int t = 0; t < 16; ++t) {
    int col = lane + 32 * t;
    float s = e[col] + e[stride + col] + e[2 * stride + col] + e[3 * stride + col];
    vals[t] = s;
    mn = fminf(mn, s);
  }
#pragma unroll
  for (int m = 16; m >= 1; m >>= 1) mn = fminf(mn, __shfl_xor(mn, m, 32));

  float s = 0.f;
#pragma unroll
  for (int t = 0; t < 16; ++t) {
    vals[t] = __expf(mn - vals[t]);
    s += vals[t];
  }
#pragma unroll
  for (int m = 16; m >= 1; m >>= 1) s += __shfl_xor(s, m, 32);

  const float inv = 1.0f / s;
  _Float16* a = attn + (size_t)row * C;
#pragma unroll
  for (int t = 0; t < 16; ++t) a[lane + 32 * t] = (_Float16)(vals[t] * inv);
}

// ---------------------------------------------------------------------------
// Kernel 4: out = gamma * (attn @ q) + x   (NT via qhT), async double-buffered
// grid (N/128, C/128, B), block 256
// ---------------------------------------------------------------------------
__global__ __launch_bounds__(256) void cam_av_wmma(
    const _Float16* __restrict__ attn, const _Float16* __restrict__ qhT,
    const float* __restrict__ x, const float* __restrict__ gamma,
    float* __restrict__ out) {
  __shared__ alignas(16) _Float16 As[2][TILE][LDSW];
  __shared__ alignas(16) _Float16 Bs[2][TILE][LDSW];

  const int b     = blockIdx.z;
  const int mBase = blockIdx.y * TILE;   // channel rows
  const int nBase = blockIdx.x * TILE;   // pixel cols
  const _Float16* Ab = attn + (size_t)b * C * C;
  const _Float16* Bb = qhT  + (size_t)b * NPIX * C;

  const int tid   = threadIdx.x;
  const int lane  = tid & 31;
  const int wave  = tid >> 5;
  const int waveM = wave & 1;
  const int waveN = wave >> 1;
  const int half  = lane >> 4;
  const int r15   = lane & 15;

  const int ldRow = tid >> 2;
  const int ldCol = (tid & 3) * 8;

  v8f acc[4][2] = {};

  auto issue = [&](int buf, int k0) {
#pragma unroll
    for (int p = 0; p < 2; ++p) {
      int row = ldRow + p * 64;
      cp_async_b128(&As[buf][row][ldCol], Ab + (size_t)(mBase + row) * C + k0 + ldCol);
      cp_async_b128(&Bs[buf][row][ldCol], Bb + (size_t)(nBase + row) * C + k0 + ldCol);
    }
  };

  issue(0, 0);
  wait_async0();
  __syncthreads();

  int cur = 0;
  for (int k0 = 0; k0 < C; k0 += KSTEP) {
    if (k0 + KSTEP < C) issue(cur ^ 1, k0 + KSTEP);

    FragU aF[4], bF[2];
#pragma unroll
    for (int i = 0; i < 4; ++i) {
      int row = waveM * 64 + i * 16 + r15;
      aF[i].h[0] = *(const h8*)(&As[cur][row][half * 8]);
      aF[i].h[1] = *(const h8*)(&As[cur][row][16 + half * 8]);
    }
#pragma unroll
    for (int j = 0; j < 2; ++j) {
      int row = waveN * 32 + j * 16 + r15;
      bF[j].h[0] = *(const h8*)(&Bs[cur][row][half * 8]);
      bF[j].h[1] = *(const h8*)(&Bs[cur][row][16 + half * 8]);
    }
#pragma unroll
    for (int i = 0; i < 4; ++i)
#pragma unroll
      for (int j = 0; j < 2; ++j)
        acc[i][j] = __builtin_amdgcn_wmma_f32_16x16x32_f16(
            false, aF[i].v, false, bF[j].v, (short)0, acc[i][j], false, false);

    wait_async0();
    __syncthreads();
    cur ^= 1;
  }

  const float g = gamma[0];
  const float* xb = x + (size_t)b * C * NPIX;
  float* ob = out + (size_t)b * C * NPIX;
#pragma unroll
  for (int i = 0; i < 4; ++i)
#pragma unroll
    for (int j = 0; j < 2; ++j) {
      int rowG = mBase + waveM * 64 + i * 16 + half * 8;
      int colG = nBase + waveN * 32 + j * 16 + r15;
      size_t base = (size_t)rowG * NPIX + colG;
#pragma unroll
      for (int v = 0; v < 8; ++v)
        ob[base + (size_t)v * NPIX] = g * acc[i][j][v] + xb[base + (size_t)v * NPIX];
    }
}

// ---------------------------------------------------------------------------
extern "C" void kernel_launch(void* const* d_in, const int* in_sizes, int n_in,
                              void* d_out, int out_size, void* d_ws, size_t ws_size,
                              hipStream_t stream) {
  const float* x     = (const float*)d_in[0];
  // d_in[1] (y) is dead code in the reference
  const float* gamma = (const float*)d_in[2];
  float* out = (float*)d_out;

  char* ws = (char*)d_ws;
  const size_t qBytes = (size_t)BATCH * C * NPIX * 2;         // 32 MB each
  const size_t pBytes = (size_t)KSPLIT * BATCH * C * C * 4;   // 32 MB
  _Float16* qh   = (_Float16*)ws;
  _Float16* qhT  = (_Float16*)(ws + qBytes);
  float*    part = (float*)   (ws + 2 * qBytes);
  _Float16* attn = (_Float16*)(ws + 2 * qBytes + pBytes);

  cam_convert<<<dim3(NPIX / 32, C / 32, BATCH), dim3(32, 8), 0, stream>>>(x, qh, qhT);
  cam_gram_wmma<<<dim3(C / TILE, C / TILE, BATCH * KSPLIT), 256, 0, stream>>>(qh, part);
  cam_softmax<<<(BATCH * C) / 8, 256, 0, stream>>>(part, attn);
  cam_av_wmma<<<dim3(NPIX / TILE, C / TILE, BATCH), 256, 0, stream>>>(attn, qhT, x, gamma, out);
}
